// LinearAttention_90512140796373
// MI455X (gfx1250) — compile-verified, matrix operand order checked by reading, not run on hardware
//
#include <hip/hip_runtime.h>
#include <hip/hip_bf16.h>

// ---------------------------------------------------------------------------
// LinearAttention for MI455X (gfx1250, wave32, WMMA)
//   B=8, N=4096 (64x64 image), C=512, HEADS=16, HD=32
//   Stage 1: qk GEMM (bf16 WMMA, fp32 accum, double-buffered LDS) + ELU+1
//   Stage 2: k_mean reduction
//   Stage 3: kv = rope(k)^T @ v / N  (bf16 WMMA)
//   Stage 4: out = (rope(q) @ kv) * z  (bf16 WMMA, K=32 exact)
//   Stage 5: out += depthwise 3x3 LePE(x) + bias  (x is L2-resident: 64MB<192MB)
// ---------------------------------------------------------------------------

typedef __attribute__((ext_vector_type(16))) __bf16 v16bf;
typedef __attribute__((ext_vector_type(8)))  __bf16 v8bf;
typedef __attribute__((ext_vector_type(4)))  __bf16 v4bf;
typedef __attribute__((ext_vector_type(8)))  float  v8f;

#define B_    8
#define N_    4096
#define C_    512
#define HEADS_ 16
#define HD_   32
#define TWO_C 1024
#define MROWS (B_ * N_)           // 32768
#define INV_N (1.0f / 4096.0f)
#define LDSTR 40                   // LDS row stride in bf16 elems (80B, conflict-free)
// theta_p = 10000^(-p/128) = exp(-p * ln(10000)/128)
#define ROPE_COEF (-0.0719557839f)

__device__ inline v8f wmma_bf16(v16bf a, v16bf b, v8f c) {
  return __builtin_amdgcn_wmma_f32_16x16x32_bf16(
      /*neg_a=*/false, a, /*neg_b=*/false, b,
      /*c_mod=*/(short)0, c, /*reuse_a=*/false, /*reuse_b=*/false);
}

// Load a 16x32 fragment (A, or B^T stored as [col][K]) from an LDS tile with
// row stride LDSTR. lane 0-15: row=lane, K base 0; lane 16-31: row=lane-16, Kbase 8.
// Element e of the v16bf holds K = (e>=8 ? 16 : 0) + kbase + (e&7).
__device__ inline v16bf ld_frag40(const __bf16* tileBase, int row, int kbase) {
  const __bf16* p = tileBase + row * LDSTR;
  v8bf lo = *(const v8bf*)(p + kbase);
  v8bf hi = *(const v8bf*)(p + 16 + kbase);
  v16bf r;
#pragma unroll
  for (int i = 0; i < 8; ++i) { r[i] = lo[i]; r[i + 8] = hi[i]; }
  return r;
}

// pack float4 -> 4 bf16 (v_cvt_pk_bf16_f32 x2) and store as one b64
__device__ inline void st4bf(__bf16* dst, float4 v) {
  v4bf o;
  o[0] = (__bf16)v.x; o[1] = (__bf16)v.y; o[2] = (__bf16)v.z; o[3] = (__bf16)v.w;
  *(v4bf*)dst = o;
}

// ---------------------------------------------------------------------------
// Kernel 0: zero scratch accumulators (kmean + kv)
// ---------------------------------------------------------------------------
__global__ void la_zero_kernel(float* __restrict__ p, int n) {
  int i = blockIdx.x * 256 + threadIdx.x;
  if (i < n) p[i] = 0.0f;
}

// ---------------------------------------------------------------------------
// Kernel 1: qk = x @ Wqk + bqk, ELU+1, split into q / k.
// 128x128 tile per 256-thread block (8 waves as 4x2), K-step 32 via WMMA bf16.
// Double-buffered LDS; next tile's global loads issue BEFORE the WMMAs so the
// load latency hides under matrix math; wait+cvt+ds_store happen after.
// ---------------------------------------------------------------------------
struct StageRegs { float4 a0, a1; float4 b[4]; };

__device__ inline StageRegs gemm_load(const float* __restrict__ x,
                                      const float* __restrict__ Wqk,
                                      int rowBase, int colBase, int k0, int t) {
  StageRegs r;
  // A: 512 float4 = 64x32 (rows 0-63 only); thread owns vec t and t+256
  int v0 = t, v1 = t + 256;
  r.a0 = *(const float4*)&x[(rowBase + (v0 >> 3)) * C_ + k0 + (v0 & 7) * 4];
  r.a1 = *(const float4*)&x[(rowBase + (v1 >> 3)) * C_ + k0 + (v1 & 7) * 4];
  // B tile 32x128 = 1024 float4 (along col); 4 per thread, coalesced
#pragma unroll
  for (int i = 0; i < 4; ++i) {
    int u = i * 256 + t;
    r.b[i] = *(const float4*)&Wqk[(k0 + (u >> 5)) * TWO_C + colBase + (u & 31) * 4];
  }
  return r;
}

__device__ inline void gemm_store(const StageRegs& r, __bf16* Asb, __bf16* Bsb,
                                  int t) {
  int v0 = t, v1 = t + 256;
  st4bf(&Asb[(v0 >> 3) * LDSTR + (v0 & 7) * 4], r.a0);
  st4bf(&Asb[(v1 >> 3) * LDSTR + (v1 & 7) * 4], r.a1);
#pragma unroll
  for (int i = 0; i < 4; ++i) {
    int u = i * 256 + t;
    int kk = u >> 5, c0 = (u & 31) * 4;
    Bsb[(c0 + 0) * LDSTR + kk] = (__bf16)r.b[i].x;   // transposed: [col][K]
    Bsb[(c0 + 1) * LDSTR + kk] = (__bf16)r.b[i].y;
    Bsb[(c0 + 2) * LDSTR + kk] = (__bf16)r.b[i].z;
    Bsb[(c0 + 3) * LDSTR + kk] = (__bf16)r.b[i].w;
  }
}

__global__ __launch_bounds__(256)
void la_qkgemm_kernel(const float* __restrict__ x, const float* __restrict__ Wqk,
                      const float* __restrict__ bqk,
                      float* __restrict__ q, float* __restrict__ k) {
  __shared__ __align__(16) __bf16 Asb[2][128 * LDSTR];   // [row][K]
  __shared__ __align__(16) __bf16 Bsb[2][128 * LDSTR];   // [col][K] (transposed)

  const int t    = threadIdx.x;
  const int lane = t & 31;
  const int wave = t >> 5;          // 0..7
  const int mW   = wave >> 1;       // 0..3  -> 32 rows each
  const int nW   = wave & 1;        // 0..1  -> 64 cols each
  const int rowBase = blockIdx.x * 128;
  const int colBase = blockIdx.y * 128;

  const v8f zero8 = {0.f, 0.f, 0.f, 0.f, 0.f, 0.f, 0.f, 0.f};
  v8f acc[2][4];
#pragma unroll
  for (int i = 0; i < 2; ++i)
#pragma unroll
    for (int j = 0; j < 4; ++j) acc[i][j] = zero8;

  const int rsel = lane & 15;
  const int kbse = (lane >> 4) * 8;

  // prologue: stage K-tile 0 into buffer 0
  {
    StageRegs sr = gemm_load(x, Wqk, rowBase, colBase, 0, t);
    gemm_store(sr, Asb[0], Bsb[0], t);
  }
  __syncthreads();

  for (int kt = 0; kt < 16; ++kt) {
    const int buf = kt & 1;
    StageRegs sr;
    const bool more = (kt + 1) < 16;
    if (more)  // issue next tile's global loads first (latency hidden by MMA)
      sr = gemm_load(x, Wqk, rowBase, colBase, (kt + 1) * 32, t);

    v16bf afr[2], bfr[4];
#pragma unroll
    for (int tm = 0; tm < 2; ++tm)
      afr[tm] = ld_frag40(&Asb[buf][(mW * 32 + tm * 16) * LDSTR], rsel, kbse);
#pragma unroll
    for (int tn = 0; tn < 4; ++tn)
      bfr[tn] = ld_frag40(&Bsb[buf][(nW * 64 + tn * 16) * LDSTR], rsel, kbse);
#pragma unroll
    for (int tm = 0; tm < 2; ++tm)
#pragma unroll
      for (int tn = 0; tn < 4; ++tn)
        acc[tm][tn] = wmma_bf16(afr[tm], bfr[tn], acc[tm][tn]);

    if (more)  // wait, convert, stash into the other buffer
      gemm_store(sr, Asb[buf ^ 1], Bsb[buf ^ 1], t);
    __syncthreads();
  }

  // epilogue: bias + ELU+1, route to q / k (branch is block-uniform)
#pragma unroll
  for (int tm = 0; tm < 2; ++tm) {
#pragma unroll
    for (int tn = 0; tn < 4; ++tn) {
      int cl = colBase + nW * 64 + tn * 16 + (lane & 15);
      float bias = bqk[cl];
      int r0 = rowBase + mW * 32 + tm * 16 + (lane >> 4) * 8;
#pragma unroll
      for (int r = 0; r < 8; ++r) {
        float v = acc[tm][tn][r] + bias;
        v = (v > 0.0f) ? (v + 1.0f) : __expf(v);     // elu(v)+1
        int row = r0 + r;
        if (cl < C_) q[row * C_ + cl] = v;
        else         k[row * C_ + (cl - C_)] = v;
      }
    }
  }
}

// ---------------------------------------------------------------------------
// Kernel 2: kmean[b][c] = sum_n k[b][n][c]   (mean applied later via 1/N)
// ---------------------------------------------------------------------------
__global__ __launch_bounds__(256)
void la_kmean_kernel(const float* __restrict__ k, float* __restrict__ kmean) {
  int b = blockIdx.x, chunk = blockIdx.y, t = threadIdx.x;
  const float* kp = k + (size_t)b * N_ * C_;
  int n0 = chunk * 256;
  float s0 = 0.f, s1 = 0.f;
#pragma unroll 8
  for (int n = n0; n < n0 + 256; ++n) {
    s0 += kp[n * C_ + t];
    s1 += kp[n * C_ + t + 256];
  }
  atomicAdd(&kmean[b * C_ + t], s0);
  atomicAdd(&kmean[b * C_ + t + 256], s1);
}

// ---------------------------------------------------------------------------
// Kernel 3: kv[b][h][d][e] = (1/N) * sum_n rope(k)[b,n,h,d] * x[b,n,h,e]
// One (b,h,n-chunk) per 128-thread block; 4 waves own the 2x2 16x16 out tiles.
// Loads hoisted: 4x float2 (k pairs) + 2x float4 (v) per thread per iter.
// ---------------------------------------------------------------------------
__global__ __launch_bounds__(128)
void la_kv_kernel(const float* __restrict__ k, const float* __restrict__ x,
                  float* __restrict__ kv) {
  __shared__ __align__(16) __bf16 Ak[HD_ * LDSTR];   // rope(k)^T : [d][token]
  __shared__ __align__(16) __bf16 Bv[HD_ * LDSTR];   // v^T       : [e][token]

  const int b = blockIdx.x, h = blockIdx.y, chunk = blockIdx.z;
  const int n0 = chunk * 512;
  const int t = threadIdx.x, lane = t & 31, wave = t >> 5;
  const int dT = wave >> 1, eT = wave & 1;
  const float* kp = k + (size_t)b * N_ * C_;
  const float* xp = x + (size_t)b * N_ * C_;

  const v8f zero8 = {0.f, 0.f, 0.f, 0.f, 0.f, 0.f, 0.f, 0.f};
  v8f acc = zero8;
  const int rsel = lane & 15;
  const int kbse = (lane >> 4) * 8;

  for (int it = 0; it < 16; ++it) {
    const int tk0 = n0 + it * 32;
    // ---- issue all global loads for this iteration ----
    float2 kpair[4];
    float4 vvec[2];
#pragma unroll
    for (int i = 0; i < 4; ++i) {                 // 512 pairs, 4 per thread
      int p = i * 128 + t;
      int tkn = p >> 4, ph = p & 15;
      kpair[i] = *(const float2*)&kp[(tk0 + tkn) * C_ + h * HD_ + 2 * ph];
    }
#pragma unroll
    for (int i = 0; i < 2; ++i) {                 // 256 float4 of v, 2 per thread
      int vec = i * 128 + t;
      int tkn = vec >> 3, e0 = (vec & 7) * 4;
      vvec[i] = *(const float4*)&xp[(tk0 + tkn) * C_ + h * HD_ + e0];
    }
    // ---- rope + convert + LDS stores ----
#pragma unroll
    for (int i = 0; i < 4; ++i) {
      int p = i * 128 + t;
      int tkn = p >> 4, ph = p & 15;
      int n = tk0 + tkn;
      int ch = h * HD_ + 2 * ph;
      int P = ch >> 1;                            // global pair idx
      float pos = (P < 128) ? (float)(n >> 6) : (float)(n & 63);
      int   pe  = (P < 128) ? P : (P - 128);
      float theta = __expf(ROPE_COEF * (float)pe);
      float sn, cs; __sincosf(pos * theta, &sn, &cs);
      float kr = kpair[i].x, ki = kpair[i].y;
      Ak[(2 * ph)     * LDSTR + tkn] = (__bf16)(kr * cs - ki * sn);
      Ak[(2 * ph + 1) * LDSTR + tkn] = (__bf16)(ki * cs + kr * sn);
    }
#pragma unroll
    for (int i = 0; i < 2; ++i) {
      int vec = i * 128 + t;
      int tkn = vec >> 3, e0 = (vec & 7) * 4;
      Bv[(e0 + 0) * LDSTR + tkn] = (__bf16)vvec[i].x;
      Bv[(e0 + 1) * LDSTR + tkn] = (__bf16)vvec[i].y;
      Bv[(e0 + 2) * LDSTR + tkn] = (__bf16)vvec[i].z;
      Bv[(e0 + 3) * LDSTR + tkn] = (__bf16)vvec[i].w;
    }
    __syncthreads();
    v16bf af = ld_frag40(&Ak[(dT * 16) * LDSTR], rsel, kbse);
    v16bf bf = ld_frag40(&Bv[(eT * 16) * LDSTR], rsel, kbse);
    acc = wmma_bf16(af, bf, acc);
    __syncthreads();
  }

  const int e  = eT * 16 + (lane & 15);
  const int d0 = dT * 16 + (lane >> 4) * 8;
#pragma unroll
  for (int r = 0; r < 8; ++r)
    atomicAdd(&kv[(((size_t)b * HEADS_ + h) * HD_ + d0 + r) * HD_ + e],
              acc[r] * INV_N);
}

// ---------------------------------------------------------------------------
// Kernel 4: out[b,n,h,:] = z * (rope(q)[b,n,h,:] @ kv[b,h])
// K=32 == one bf16 WMMA depth: single MMA pair per 16-token tile.
// ---------------------------------------------------------------------------
__global__ __launch_bounds__(128)
void la_attn_kernel(const float* __restrict__ q, const float* __restrict__ kmean,
                    const float* __restrict__ kv, float* __restrict__ out) {
  __shared__ __align__(16) __bf16 Aq[64 * LDSTR];    // rope(q): [token][d]
  __shared__ __align__(16) __bf16 Bkv[HD_ * LDSTR];  // kv^T   : [e][d]
  __shared__ float km[HD_];
  __shared__ float zl[64];

  const int b = blockIdx.x, h = blockIdx.y, tile = blockIdx.z;
  const int n0 = tile * 64;
  const int t = threadIdx.x, lane = t & 31, wave = t >> 5;  // 4 waves
  const float* qp = q + (size_t)b * N_ * C_;

  // ---- issue all global loads up front ----
  float kmv = (t < HD_) ? kmean[b * C_ + h * HD_ + t] : 0.f;
  float4 kvvec[2];
#pragma unroll
  for (int i = 0; i < 2; ++i) {                    // kv: 256 float4, 2 per thread
    int vec = i * 128 + t;
    int d = vec >> 3, e0 = (vec & 7) * 4;
    kvvec[i] = *(const float4*)&kv[(((size_t)b * HEADS_ + h) * HD_ + d) * HD_ + e0];
  }
  float2 qpair[8];
#pragma unroll
  for (int i = 0; i < 8; ++i) {                    // 1024 pairs, 8 per thread
    int p = i * 128 + t;
    int tkn = p >> 4, ph = p & 15;
    qpair[i] = *(const float2*)&qp[(n0 + tkn) * C_ + h * HD_ + 2 * ph];
  }

  // ---- convert + LDS stores ----
  if (t < HD_) km[t] = kmv;
#pragma unroll
  for (int i = 0; i < 2; ++i) {
    int vec = i * 128 + t;
    int d = vec >> 3, e0 = (vec & 7) * 4;
    Bkv[(e0 + 0) * LDSTR + d] = (__bf16)kvvec[i].x;   // [e][d]
    Bkv[(e0 + 1) * LDSTR + d] = (__bf16)kvvec[i].y;
    Bkv[(e0 + 2) * LDSTR + d] = (__bf16)kvvec[i].z;
    Bkv[(e0 + 3) * LDSTR + d] = (__bf16)kvvec[i].w;
  }
#pragma unroll
  for (int i = 0; i < 8; ++i) {
    int p = i * 128 + t;
    int tkn = p >> 4, ph = p & 15;
    int n = n0 + tkn;
    int ch = h * HD_ + 2 * ph;
    int P = ch >> 1;
    float pos = (P < 128) ? (float)(n >> 6) : (float)(n & 63);
    int   pe  = (P < 128) ? P : (P - 128);
    float theta = __expf(ROPE_COEF * (float)pe);
    float sn, cs; __sincosf(pos * theta, &sn, &cs);
    float qr = qpair[i].x, qi = qpair[i].y;
    Aq[tkn * LDSTR + 2 * ph]     = (__bf16)(qr * cs - qi * sn);
    Aq[tkn * LDSTR + 2 * ph + 1] = (__bf16)(qi * cs + qr * sn);
  }
  __syncthreads();

  // z = 1 / (q . k_mean + eps)  (vectorized dot, km ready after barrier)
  if (t < 64) {
    const float4* qv = (const float4*)&qp[(n0 + t) * C_ + h * HD_];
    float dot = 0.f;
#pragma unroll
    for (int i = 0; i < 8; ++i) {
      float4 qq = qv[i];
      dot += qq.x * km[i * 4 + 0] + qq.y * km[i * 4 + 1] +
             qq.z * km[i * 4 + 2] + qq.w * km[i * 4 + 3];
    }
    zl[t] = 1.0f / (dot * INV_N + 1e-6f);
  }
  __syncthreads();

  const int rsel = lane & 15;
  const int kbse = (lane >> 4) * 8;
  v16bf af = ld_frag40(&Aq[(wave * 16) * LDSTR], rsel, kbse);
  v16bf b0 = ld_frag40(&Bkv[0], rsel, kbse);
  v16bf b1 = ld_frag40(&Bkv[16 * LDSTR], rsel, kbse);
  const v8f zero8 = {0.f, 0.f, 0.f, 0.f, 0.f, 0.f, 0.f, 0.f};
  v8f acc0 = wmma_bf16(af, b0, zero8);
  v8f acc1 = wmma_bf16(af, b1, zero8);

  const int col  = lane & 15;
  const int tok0 = wave * 16 + (lane >> 4) * 8;
#pragma unroll
  for (int r = 0; r < 8; ++r) {
    int tok = tok0 + r;
    float zz = zl[tok];
    size_t base = ((size_t)b * N_ + (n0 + tok)) * C_ + h * HD_;
    out[base + col]      = acc0[r] * zz;
    out[base + 16 + col] = acc1[r] * zz;
  }
}

// ---------------------------------------------------------------------------
// Kernel 5: out += depthwise 3x3 conv(x) + bias   (x is L2-resident)
// ---------------------------------------------------------------------------
__global__ __launch_bounds__(256)
void la_lepe_kernel(const float* __restrict__ x, const float* __restrict__ lw,
                    const float* __restrict__ lb, float* __restrict__ out) {
  size_t idx = (size_t)blockIdx.x * 256 + threadIdx.x;   // B*N*C threads
  int c  = (int)(idx & (C_ - 1));
  size_t nc = idx >> 9;
  int n  = (int)(nc & (N_ - 1));
  int b  = (int)(nc >> 12);
  int r  = n >> 6, cc = n & 63;
  float a = lb[c];
#pragma unroll
  for (int ky = -1; ky <= 1; ++ky) {
    int rr = r + ky;
    if ((unsigned)rr >= 64u) continue;
#pragma unroll
    for (int kx = -1; kx <= 1; ++kx) {
      int cx = cc + kx;
      if ((unsigned)cx >= 64u) continue;
      a += lw[c * 9 + (ky + 1) * 3 + (kx + 1)] *
           x[((size_t)b * N_ + (rr * 64 + cx)) * C_ + c];
    }
  }
  out[idx] += a;
}

// ---------------------------------------------------------------------------
extern "C" void kernel_launch(void* const* d_in, const int* in_sizes, int n_in,
                              void* d_out, int out_size, void* d_ws, size_t ws_size,
                              hipStream_t stream) {
  const float* x    = (const float*)d_in[0];
  // d_in[1]=h, d_in[2]=w are fixed (64,64) per the reference constants
  const float* Wqk  = (const float*)d_in[3];
  const float* bqk  = (const float*)d_in[4];
  const float* lw   = (const float*)d_in[5];
  const float* lb   = (const float*)d_in[6];
  float* out = (float*)d_out;

  float* ws    = (float*)d_ws;
  float* q     = ws;                                   // 16,777,216 f
  float* k     = q + (size_t)MROWS * C_;               // 16,777,216 f
  float* kmean = k + (size_t)MROWS * C_;               // 4,096 f
  float* kv    = kmean + (size_t)B_ * C_;              // 131,072 f

  const int accN = B_ * C_ + B_ * HEADS_ * HD_ * HD_;  // kmean + kv
  la_zero_kernel<<<(accN + 255) / 256, 256, 0, stream>>>(kmean, accN);

  la_qkgemm_kernel<<<dim3(MROWS / 128, TWO_C / 128), 256, 0, stream>>>(
      x, Wqk, bqk, q, k);
  la_kmean_kernel<<<dim3(B_, 16), 256, 0, stream>>>(k, kmean);
  la_kv_kernel<<<dim3(B_, HEADS_, 8), 128, 0, stream>>>(k, x, kv);
  la_attn_kernel<<<dim3(B_, HEADS_, N_ / 64), 128, 0, stream>>>(q, kmean, kv, out);
  la_lepe_kernel<<<(unsigned)((size_t)B_ * N_ * C_ / 256), 256, 0, stream>>>(
      x, lw, lb, out);
}
